// SwinTransformerBlock_77223511982374
// MI455X (gfx1250) — matrix operand order repaired and checked by reading, not compiled, not run on hardware
//
#include <hip/hip_runtime.h>

typedef __attribute__((ext_vector_type(16))) _Float16 v16h;
typedef __attribute__((ext_vector_type(8)))  float    v8f;

#define WMMA_F16(a, b, c) \
  __builtin_amdgcn_wmma_f32_16x16x32_f16(false, (a), false, (b), (short)0, (c), false, false)

constexpr int   cB     = 32;
constexpr int   cH     = 56;
constexpr int   cW     = 56;
constexpr int   cC     = 256;
constexpr int   cHEADS = 8;
constexpr int   cD     = 32;     // head dim
constexpr int   cWS    = 7;
constexpr int   cSS    = 3;
constexpr int   cN     = 49;     // tokens per window
constexpr int   cNW    = 64;     // windows per image
constexpr int   cWIN   = cB * cNW;          // 2048
constexpr long  cTOK   = (long)cWIN * cN;   // 100352
constexpr int   cMLP   = 1024;
constexpr float cSCALE = 0.17677669529663689f;  // (C/HEADS)^-0.5
constexpr int   cKC    = 256;    // K-chunk staged in LDS per TDM op (halves)

// ---------------------------------------------------------------------------
// Fragment loaders (wave32, 16x16x32 f16 WMMA; contiguous k order used for both A and B, not the ISA table order)
// A 16x32 (MxK), row-major source, leading dim `ld` in halves:
//   lane: m = lane&15, half = lane>>4
//   VGPR v: K = 16*half + 2v + pair  (contiguous 32B per lane)
// B 32x16 (KxN), loaded from Bt (N x K row-major): lane column n = lane&15,
//   VGPR v: K = 16*half + 2v + pair  (contiguous 32B per lane)
// C/D 16x16 f32: lane n = lane&15, vgpr r: row m = r + 8*(lane>>4)
// ---------------------------------------------------------------------------

__device__ __forceinline__ v16h load_a_frag(const _Float16* __restrict__ p,
                                            long row0, int k0, int ld) {
  const int lane = threadIdx.x & 31;
  const int m = lane & 15, half = lane >> 4;
  const unsigned int* rp =
      (const unsigned int*)(p + (row0 + m) * (long)ld + k0 + half * 16);
  union { v16h h; unsigned int u[8]; } r;
#pragma unroll
  for (int d = 0; d < 8; ++d) r.u[d] = rp[d];
  return r.h;
}

__device__ __forceinline__ v16h load_a_frag_pad(const _Float16* __restrict__ p,
                                                int row0, int k0, int ld, int maxRow) {
  const int lane = threadIdx.x & 31;
  const int m = lane & 15, half = lane >> 4;
  union { v16h h; unsigned int u[8]; } r;
  if (row0 + m < maxRow) {
    const unsigned int* rp =
        (const unsigned int*)(p + (long)(row0 + m) * ld + k0 + half * 16);
#pragma unroll
    for (int d = 0; d < 8; ++d) r.u[d] = rp[d];
  } else {
#pragma unroll
    for (int d = 0; d < 8; ++d) r.u[d] = 0u;
  }
  return r.h;
}

__device__ __forceinline__ v16h load_b_frag(const _Float16* __restrict__ bt,
                                            long col0, int k0, int ld) {
  const int lane = threadIdx.x & 31;
  const int nn = lane & 15, half = lane >> 4;
  const unsigned int* rp =
      (const unsigned int*)(bt + (col0 + nn) * (long)ld + k0 + half * 16);
  union { v16h h; unsigned int u[8]; } r;
#pragma unroll
  for (int d = 0; d < 8; ++d) r.u[d] = rp[d];
  return r.h;
}

__device__ __forceinline__ v16h load_b_frag_pad(const _Float16* __restrict__ bt,
                                                int col0, int k0, int ld, int maxRow) {
  const int lane = threadIdx.x & 31;
  const int nn = lane & 15, half = lane >> 4;
  union { v16h h; unsigned int u[8]; } r;
  if (col0 + nn < maxRow) {
    const unsigned int* rp =
        (const unsigned int*)(bt + (long)(col0 + nn) * ld + k0 + half * 16);
#pragma unroll
    for (int d = 0; d < 8; ++d) r.u[d] = rp[d];
  } else {
#pragma unroll
    for (int d = 0; d < 8; ++d) r.u[d] = 0u;
  }
  return r.h;
}

// ---------------------------------------------------------------------------
// Tensor Data Mover: stage a 2-D f16 tile (tileX halves x tileY rows, row
// stride strideX halves) from global into LDS.  D# layout per ISA ch.8.
// Toolchain probe: 6-arg builtin (clang-23 / therock-10.0 form).
// ---------------------------------------------------------------------------
#if __has_builtin(__builtin_amdgcn_tensor_load_to_lds)
#define CDNA5_TDM 1
typedef unsigned int v4u __attribute__((ext_vector_type(4)));
typedef int          v4i __attribute__((ext_vector_type(4)));
typedef int          v8i __attribute__((ext_vector_type(8)));

__device__ __forceinline__ void tdm_load_tile(unsigned lds_off,
                                              const _Float16* g,
                                              int tileX, int tileY, int strideX) {
  unsigned long long ga = (unsigned long long)(const void*)g;
  // group0: count=1 | lds_addr | global_addr[56:0] | type=2 (bits 127:126)
  v4u g0 = {1u, lds_off, (unsigned)ga,
            (unsigned)((ga >> 32) & 0x01FFFFFFu) | 0x80000000u};
  // group1: data_size=2B; tensor_dim0/1 == tile_dim0/1; dim0 stride = strideX
  v8i g1 = {(int)(1u << 16),
            (int)((unsigned)(tileX & 0xFFFF) << 16),
            (int)(((unsigned)tileX >> 16) | ((unsigned)(tileY & 0xFFFF) << 16)),
            (int)(((unsigned)tileY >> 16) | ((unsigned)(tileX & 0xFFFF) << 16)),
            (int)(tileY & 0xFFFF),
            strideX, 0, 0};
  v4i gz4 = {0, 0, 0, 0};
  v8i gz8 = {0, 0, 0, 0, 0, 0, 0, 0};
  __builtin_amdgcn_tensor_load_to_lds(g0, g1, gz4, gz4, gz8, 0);
}
#endif

// ---------------------------------------------------------------------------
// Block GEMM core: 4 waves, block computes 64 rows x 64 cols.  B tile for the
// block's 64 columns is staged in LDS (TDM), A streams from global.  Each wave
// holds 4 f32 accumulators (16x64 strip), A fragment reused 4x per K-step.
// ---------------------------------------------------------------------------
template <int K>
__device__ __forceinline__ void gemm_block64(const _Float16* __restrict__ A,
                                             const _Float16* __restrict__ Bt,
                                             long row0, int col0,
                                             _Float16* Bs, v8f acc[4]) {
#pragma unroll 1
  for (int kc = 0; kc < K; kc += cKC) {
    if (kc) __syncthreads();
#if defined(CDNA5_TDM)
    if (threadIdx.x < 32) {
      unsigned ldso = (unsigned)(unsigned long long)(void*)Bs;
      tdm_load_tile(ldso, Bt + (long)col0 * K + kc, cKC, 64, K);
      __builtin_amdgcn_s_wait_tensorcnt(0);
    }
#else
    for (int i = threadIdx.x; i < (64 * cKC) / 8; i += 128) {
      int rr = i >> 5, rc = (i & 31) * 8;
      *(uint4*)(Bs + rr * cKC + rc) =
          *(const uint4*)(Bt + (long)(col0 + rr) * K + kc + rc);
    }
#endif
    __syncthreads();
#pragma unroll
    for (int k0 = 0; k0 < cKC; k0 += 32) {
      v16h a = load_a_frag(A, row0, kc + k0, K);
#pragma unroll
      for (int c = 0; c < 4; ++c) {
        v16h b = load_b_frag(Bs, c * 16, k0, cKC);
        acc[c] = WMMA_F16(a, b, acc[c]);
      }
    }
  }
}

// ---------------------------------------------------------------------------
// Weight transpose + f32 -> f16:  in (K,N) row-major  ->  out (N,K) row-major
// ---------------------------------------------------------------------------
__global__ void k_w_t(const float* __restrict__ in, _Float16* __restrict__ out,
                      int K, int N) {
  int i = blockIdx.x * blockDim.x + threadIdx.x;
  if (i >= K * N) return;
  int n = i / K, kk = i % K;
  out[i] = (_Float16)in[(long)kk * N + n];
}

// ---------------------------------------------------------------------------
// LayerNorm (one wave per token).  remap=1: fused cyclic shift + window
// partition (output token order = window-token order of the shifted image).
// ---------------------------------------------------------------------------
__global__ __launch_bounds__(128) void k_ln(const float* __restrict__ x,
                                            const float* __restrict__ g,
                                            const float* __restrict__ bb,
                                            _Float16* __restrict__ o16,
                                            int remap) {
  long t = (long)blockIdx.x * 4 + (threadIdx.x >> 5);
  int lane = threadIdx.x & 31;
  long src = t;
  if (remap) {
    int w = (int)(t / cN), p = (int)(t % cN);
    int bi = w >> 6, wi = w & 63;
    int rs = (wi >> 3) * cWS + p / cWS;
    int cs = (wi & 7) * cWS + p % cWS;
    int ro = rs + cSS; if (ro >= cH) ro -= cH;
    int co = cs + cSS; if (co >= cW) co -= cW;
    src = (long)bi * (cH * cW) + ro * cW + co;
  }
  const float* xp = x + src * cC;
  float vals[8], s = 0.f, s2 = 0.f;
#pragma unroll
  for (int j = 0; j < 8; ++j) {
    float v = xp[lane + 32 * j];
    vals[j] = v; s += v; s2 += v * v;
  }
#pragma unroll
  for (int m = 1; m < 32; m <<= 1) {
    s  += __shfl_xor(s,  m, 32);
    s2 += __shfl_xor(s2, m, 32);
  }
  float mean = s * (1.0f / cC);
  float var  = s2 * (1.0f / cC) - mean * mean;
  float inv  = rsqrtf(var + 1e-5f);
#pragma unroll
  for (int j = 0; j < 8; ++j) {
    int c = lane + 32 * j;
    o16[t * cC + c] = (_Float16)((vals[j] - mean) * inv * g[c] + bb[c]);
  }
}

// ---------------------------------------------------------------------------
// QKV GEMM: (100352 x 256) x (256 x 768); scatters q/k/v[win][head][tok][dim]
// (f16), q pre-scaled by SCALE, bias fused.
// ---------------------------------------------------------------------------
__global__ __launch_bounds__(128) void k_gemm_qkv(const _Float16* __restrict__ A,
                                                  const _Float16* __restrict__ Bt,
                                                  const float* __restrict__ bias,
                                                  _Float16* __restrict__ qb,
                                                  _Float16* __restrict__ kb,
                                                  _Float16* __restrict__ vb) {
  __shared__ _Float16 Bs[64 * cKC];
  const int NT = (3 * cC) / 64;  // 12
  int bm = blockIdx.x / NT, bn = blockIdx.x % NT;
  long row0 = (long)bm * 64 + (threadIdx.x >> 5) * 16;
  int col0 = bn * 64;
  v8f acc[4] = {};
  gemm_block64<cC>(A, Bt, row0, col0, Bs, acc);
  int lane = threadIdx.x & 31, n = lane & 15, half = lane >> 4;
#pragma unroll
  for (int c = 0; c < 4; ++c) {
    int cg = col0 + c * 16 + n;
    float bs = bias[cg];
    int part = cg >> 8, rem = cg & 255, head = rem >> 5, dim = rem & 31;
    _Float16* dst = (part == 0) ? qb : ((part == 1) ? kb : vb);
    float scale = (part == 0) ? cSCALE : 1.0f;
#pragma unroll
    for (int r = 0; r < 8; ++r) {
      long t = row0 + r + 8 * half;
      int ww = (int)(t / cN), p = (int)(t % cN);
      dst[(((long)(ww * cHEADS + head)) * cN + p) * cD + dim] =
          (_Float16)((acc[c][r] + bs) * scale);
    }
  }
}

// ---------------------------------------------------------------------------
// Windowed attention: one wave per (window, head); N=49 padded to 64.
// ---------------------------------------------------------------------------
__device__ __forceinline__ int region56(int g) {
  return g < (cH - cWS) ? 0 : (g < (cH - cSS) ? 1 : 2);
}

__global__ __launch_bounds__(128) void k_attn(const _Float16* __restrict__ qg,
                                              const _Float16* __restrict__ kg,
                                              const _Float16* __restrict__ vg,
                                              const float* __restrict__ rel_bias,
                                              _Float16* __restrict__ aout) {
  __shared__ _Float16 Sb[4][64 * 64];
  const int wslot = threadIdx.x >> 5;
  const int u = blockIdx.x * 4 + wslot;
  const int w = u >> 3, head = u & 7;
  const int lane = threadIdx.x & 31;
  const int n = lane & 15, half = lane >> 4;
  const long base = ((long)(w * cHEADS + head)) * cN * cD;
  const _Float16* qp = qg + base;
  const _Float16* kp = kg + base;
  const _Float16* vp = vg + base;
  const int wi = w & 63, wr = wi >> 3, wc = wi & 7;
  _Float16* Sm = Sb[wslot];

  v16h Bk[4];
#pragma unroll
  for (int ct = 0; ct < 4; ++ct) Bk[ct] = load_b_frag_pad(kp, ct * 16, 0, cD, cN);

#pragma unroll
  for (int rt = 0; rt < 4; ++rt) {
    v16h Aq = load_a_frag_pad(qp, rt * 16, 0, cD, cN);
    v8f st[4];
#pragma unroll
    for (int ct = 0; ct < 4; ++ct) { v8f z = {}; st[ct] = WMMA_F16(Aq, Bk[ct], z); }

    // relative-position bias + shifted-window mask (computed arithmetically)
#pragma unroll
    for (int ct = 0; ct < 4; ++ct) {
      int col = ct * 16 + n;
      int qr = col / cWS, qc = col % cWS;
#pragma unroll
      for (int r = 0; r < 8; ++r) {
        int row = rt * 16 + r + 8 * half;
        float sv = st[ct][r];
        if (row < cN && col < cN) {
          int pr = row / cWS, pc = row % cWS;
          int ridx = (pr - qr + cWS - 1) * (2 * cWS - 1) + (pc - qc + cWS - 1);
          sv += rel_bias[ridx * cHEADS + head];
          int labp = 3 * region56(wr * cWS + pr) + region56(wc * cWS + pc);
          int labq = 3 * region56(wr * cWS + qr) + region56(wc * cWS + qc);
          if (labp != labq) sv -= 100.0f;
        } else {
          sv = -3.0e38f;
        }
        st[ct][r] = sv;
      }
    }

    // softmax per row (row lives on 16 lanes of one half)
#pragma unroll
    for (int r = 0; r < 8; ++r) {
      float m = fmaxf(fmaxf(st[0][r], st[1][r]), fmaxf(st[2][r], st[3][r]));
#pragma unroll
      for (int msk = 1; msk < 16; msk <<= 1) m = fmaxf(m, __shfl_xor(m, msk, 32));
      float e[4]; float s = 0.f;
#pragma unroll
      for (int ct = 0; ct < 4; ++ct) {
        int col = ct * 16 + n;
        float ev = (col < cN) ? __expf(st[ct][r] - m) : 0.f;
        e[ct] = ev; s += ev;
      }
#pragma unroll
      for (int msk = 1; msk < 16; msk <<= 1) s += __shfl_xor(s, msk, 32);
      float inv = (s > 0.f) ? (1.0f / s) : 0.f;
      int row = rt * 16 + r + 8 * half;
#pragma unroll
      for (int ct = 0; ct < 4; ++ct)
        Sm[row * 64 + ct * 16 + n] = (_Float16)(e[ct] * inv);
    }
  }
  __syncthreads();

  // V as B fragments (gather: K index = token, stride cD halves)
  v16h Bv[2][2];
#pragma unroll
  for (int kt = 0; kt < 2; ++kt) {
#pragma unroll
    for (int cv = 0; cv < 2; ++cv) {
      int dim = cv * 16 + n;
      union { v16h h; _Float16 f[16]; } t;
#pragma unroll
      for (int e = 0; e < 16; ++e) {
        int kk = kt * 32 + half * 16 + e;
        t.f[e] = (kk < cN) ? vp[kk * cD + dim] : (_Float16)0.0f;
      }
      Bv[kt][cv] = t.h;
    }
  }

  // O = P x V, store (win-token, head*32+dim) f16
#pragma unroll
  for (int rt = 0; rt < 4; ++rt) {
    v8f o0 = {}, o1 = {};
#pragma unroll
    for (int kt = 0; kt < 2; ++kt) {
      v16h Ap = load_a_frag(Sm, rt * 16, kt * 32, 64);
      o0 = WMMA_F16(Ap, Bv[kt][0], o0);
      o1 = WMMA_F16(Ap, Bv[kt][1], o1);
    }
#pragma unroll
    for (int r = 0; r < 8; ++r) {
      int row = rt * 16 + r + 8 * half;
      if (row < cN) {
        long t = (long)w * cN + row;
        aout[t * cC + head * cD + n]      = (_Float16)o0[r];
        aout[t * cC + head * cD + 16 + n] = (_Float16)o1[r];
      }
    }
  }
}

// ---------------------------------------------------------------------------
// Proj GEMM + window-reverse + reverse cyclic shift + shortcut residual -> x1
// ---------------------------------------------------------------------------
__global__ __launch_bounds__(128) void k_gemm_proj(const _Float16* __restrict__ A,
                                                   const _Float16* __restrict__ Bt,
                                                   const float* __restrict__ bias,
                                                   const float* __restrict__ xin,
                                                   float* __restrict__ out) {
  __shared__ _Float16 Bs[64 * cKC];
  const int NT = cC / 64;  // 4
  int bm = blockIdx.x / NT, bn = blockIdx.x % NT;
  long row0 = (long)bm * 64 + (threadIdx.x >> 5) * 16;
  int col0 = bn * 64;
  v8f acc[4] = {};
  gemm_block64<cC>(A, Bt, row0, col0, Bs, acc);
  int lane = threadIdx.x & 31, n = lane & 15, half = lane >> 4;
#pragma unroll
  for (int r = 0; r < 8; ++r) {
    long t = row0 + r + 8 * half;
    int w = (int)(t / cN), p = (int)(t % cN);
    int bi = w >> 6, wi = w & 63;
    int rs = (wi >> 3) * cWS + p / cWS;
    int cs = (wi & 7) * cWS + p % cWS;
    int ro = rs + cSS; if (ro >= cH) ro -= cH;
    int co = cs + cSS; if (co >= cW) co -= cW;
    long tok = (long)bi * (cH * cW) + ro * cW + co;
#pragma unroll
    for (int c = 0; c < 4; ++c) {
      int cg = col0 + c * 16 + n;
      out[tok * cC + cg] = xin[tok * cC + cg] + acc[c][r] + bias[cg];
    }
  }
}

// ---------------------------------------------------------------------------
// MLP1 GEMM + bias + exact GELU -> hidden f16
// ---------------------------------------------------------------------------
__global__ __launch_bounds__(128) void k_gemm_mlp1(const _Float16* __restrict__ A,
                                                   const _Float16* __restrict__ Bt,
                                                   const float* __restrict__ bias,
                                                   _Float16* __restrict__ hid) {
  __shared__ _Float16 Bs[64 * cKC];
  const int NT = cMLP / 64;  // 16
  int bm = blockIdx.x / NT, bn = blockIdx.x % NT;
  long row0 = (long)bm * 64 + (threadIdx.x >> 5) * 16;
  int col0 = bn * 64;
  v8f acc[4] = {};
  gemm_block64<cC>(A, Bt, row0, col0, Bs, acc);
  int lane = threadIdx.x & 31, n = lane & 15, half = lane >> 4;
#pragma unroll
  for (int c = 0; c < 4; ++c) {
    int cg = col0 + c * 16 + n;
    float bs = bias[cg];
#pragma unroll
    for (int r = 0; r < 8; ++r) {
      long t = row0 + r + 8 * half;
      float z = acc[c][r] + bs;
      float gz = 0.5f * z * (1.0f + erff(z * 0.70710678118654752f));
      hid[t * cMLP + cg] = (_Float16)gz;
    }
  }
}

// ---------------------------------------------------------------------------
// MLP2 GEMM + bias + final residual (out holds x1, updated in place)
// ---------------------------------------------------------------------------
__global__ __launch_bounds__(128) void k_gemm_mlp2(const _Float16* __restrict__ A,
                                                   const _Float16* __restrict__ Bt,
                                                   const float* __restrict__ bias,
                                                   float* __restrict__ out) {
  __shared__ _Float16 Bs[64 * cKC];
  const int NT = cC / 64;  // 4
  int bm = blockIdx.x / NT, bn = blockIdx.x % NT;
  long row0 = (long)bm * 64 + (threadIdx.x >> 5) * 16;
  int col0 = bn * 64;
  v8f acc[4] = {};
  gemm_block64<cMLP>(A, Bt, row0, col0, Bs, acc);
  int lane = threadIdx.x & 31, n = lane & 15, half = lane >> 4;
#pragma unroll
  for (int c = 0; c < 4; ++c) {
    int cg = col0 + c * 16 + n;
    float bs = bias[cg];
#pragma unroll
    for (int r = 0; r < 8; ++r) {
      long t = row0 + r + 8 * half;
      long idx = t * cC + cg;
      out[idx] = out[idx] + acc[c][r] + bs;
    }
  }
}

// ---------------------------------------------------------------------------
extern "C" void kernel_launch(void* const* d_in, const int* in_sizes, int n_in,
                              void* d_out, int out_size, void* d_ws, size_t ws_size,
                              hipStream_t stream) {
  (void)in_sizes; (void)n_in; (void)out_size; (void)ws_size;

  const float* x      = (const float*)d_in[0];
  const float* ln1_g  = (const float*)d_in[1];
  const float* ln1_b  = (const float*)d_in[2];
  const float* qkv_w  = (const float*)d_in[3];
  const float* qkv_b  = (const float*)d_in[4];
  const float* rel_b  = (const float*)d_in[5];
  const float* proj_w = (const float*)d_in[6];
  const float* proj_b = (const float*)d_in[7];
  const float* ln2_g  = (const float*)d_in[8];
  const float* ln2_b  = (const float*)d_in[9];
  const float* w1     = (const float*)d_in[10];
  const float* b1     = (const float*)d_in[11];
  const float* w2     = (const float*)d_in[12];
  const float* b2     = (const float*)d_in[13];
  float* out = (float*)d_out;

  const long S = cTOK * cC;  // 25,690,112 halves per activation buffer
  _Float16* ws16 = (_Float16*)d_ws;
  _Float16* h16   = ws16;            // region0: LN1 out; later LN2 out
  _Float16* qb    = ws16 + 1 * S;
  _Float16* kb    = ws16 + 2 * S;
  _Float16* vb    = ws16 + 3 * S;
  _Float16* aout  = ws16 + 4 * S;
  _Float16* hid   = ws16 + 1 * S;    // reuses dead q/k/v/aout (4S = TOK*1024)
  _Float16* qkvT  = ws16 + 5 * S;
  _Float16* projT = qkvT + 768 * 256;
  _Float16* w1T   = projT + 256 * 256;
  _Float16* w2T   = w1T + 1024 * 256;

  const int MT64 = (int)(cTOK / 64);  // 1568

  // weight transpose/convert (f32 (K,N) -> f16 (N,K))
  k_w_t<<<(768 * 256 + 255) / 256, 256, 0, stream>>>(qkv_w, qkvT, 256, 768);
  k_w_t<<<(256 * 256 + 255) / 256, 256, 0, stream>>>(proj_w, projT, 256, 256);
  k_w_t<<<(1024 * 256 + 255) / 256, 256, 0, stream>>>(w1, w1T, 256, 1024);
  k_w_t<<<(1024 * 256 + 255) / 256, 256, 0, stream>>>(w2, w2T, 1024, 256);

  // LN1 + shift + window partition
  k_ln<<<(int)(cTOK / 4), 128, 0, stream>>>(x, ln1_g, ln1_b, h16, 1);
  // QKV
  k_gemm_qkv<<<MT64 * 12, 128, 0, stream>>>(h16, qkvT, qkv_b, qb, kb, vb);
  // attention
  k_attn<<<(cWIN * cHEADS) / 4, 128, 0, stream>>>(qb, kb, vb, rel_b, aout);
  // proj + reverse windows/shift + residual -> out (x1)
  k_gemm_proj<<<MT64 * 4, 128, 0, stream>>>(aout, projT, proj_b, x, out);
  // LN2
  k_ln<<<(int)(cTOK / 4), 128, 0, stream>>>(out, ln2_g, ln2_b, h16, 0);
  // MLP
  k_gemm_mlp1<<<MT64 * 16, 128, 0, stream>>>(h16, w1T, b1, hid);
  k_gemm_mlp2<<<MT64 * 4, 128, 0, stream>>>(hid, w2T, b2, out);
}
